// GNN_Dense_50002009260729
// MI455X (gfx1250) — compile-verified, matrix operand order checked by reading, not run on hardware
//
#include <hip/hip_runtime.h>
#include <hip/hip_bf16.h>

typedef __attribute__((ext_vector_type(16))) _Float16 v16h;
typedef __attribute__((ext_vector_type(8)))  _Float16 v8h;
typedef __attribute__((ext_vector_type(8)))  float    v8f;

#define IN_DIM 128
#define OUT_DIM 64
#define NEG_SLOPE 0.2f
#define TILE_M 32        // rows of h per block (2 row-tiles x 4 col-tiles = 8 waves)
#define LDS_K (IN_DIM + 8)   // padded f16 row stride: 272 B -> conflict-free frag loads

// Build a 16-f16 WMMA fragment half-pair from two contiguous 16 B LDS chunks.
__device__ __forceinline__ v16h frag_ld(const _Float16* p) {
    v8h lo = *(const v8h*)(p);        // K = kh .. kh+7       (ds_load_b128)
    v8h hi = *(const v8h*)(p + 16);   // K = 16+kh .. 16+kh+7 (ds_load_b128)
    return __builtin_shufflevector(lo, hi, 0, 1, 2, 3, 4, 5, 6, 7,
                                           8, 9, 10, 11, 12, 13, 14, 15);
}

// ---------------------------------------------------------------------------
// h = x @ W  (x: [n,128] f32, W: [128,64] f32) via v_wmma_f32_16x16x32_f16.
// x-tile staged as f16 rows, W staged TRANSPOSED (sWT[col][k]) so both A and B
// fragments are contiguous ds_load_b128 pairs; padded stride -> no bank
// conflicts. h is padded to a TILE_M multiple by the caller, so the store
// epilogue is 8 unconditional global_store_b32 (no exec-mask juggling).
// ---------------------------------------------------------------------------
__global__ __launch_bounds__(256) void k_gemm_h(const float* __restrict__ x,
                                                const float* __restrict__ w,
                                                float* __restrict__ h, int n) {
    __shared__ _Float16 sX[TILE_M * LDS_K];    // 8.5 KB
    __shared__ _Float16 sWT[OUT_DIM * LDS_K];  // 17 KB
    const int t = threadIdx.x;
    const int row0 = blockIdx.x * TILE_M;

    // stage W transposed: sWT[col*LDS_K + k] = w[k*64 + col]
    for (int i = t; i < OUT_DIM * IN_DIM; i += 256) {
        int col = i >> 7, k = i & 127;                 // consecutive k -> LDS coalesced
        sWT[col * LDS_K + k] = (_Float16)w[k * OUT_DIM + col];
    }
    // stage x tile (input reads clamped; padded rows of h absorb the dups)
    for (int i = t; i < TILE_M * IN_DIM; i += 256) {
        int r = i >> 7, c = i & 127;
        int gr = row0 + r; if (gr >= n) gr = n - 1;
        sX[r * LDS_K + c] = (_Float16)x[gr * IN_DIM + c];
    }
    __syncthreads();

    const int lane = t & 31, wid = t >> 5;
    const int rt = (wid >> 2) * 16;      // row-tile offset within block tile
    const int ct = (wid & 3) * 16;       // col-tile offset (0..48)
    const int mn = lane & 15;            // M for A-frag, N for B/C/D frags
    const int kh = (lane < 16) ? 0 : 8;  // K-half per ISA 16-bit fragment layout

    const _Float16* pa = &sX[(rt + mn) * LDS_K + kh];
    const _Float16* pb = &sWT[(ct + mn) * LDS_K + kh];

    v8f acc = {};
#pragma unroll
    for (int kk = 0; kk < IN_DIM; kk += 32) {
        v16h a = frag_ld(pa + kk);
        v16h b = frag_ld(pb + kk);
        acc = __builtin_amdgcn_wmma_f32_16x16x32_f16(
            /*neg_a=*/false, a, /*neg_b=*/false, b,
            /*c_mod=*/(short)0, acc, /*reuse_a=*/false, /*reuse_b=*/false);
    }

    // C/D layout: M = v + (lane<16 ? 0 : 8), N = lane&15; stores unguarded
    float* hp = h + (long long)(row0 + rt + ((lane < 16) ? 0 : 8)) * OUT_DIM + ct + mn;
#pragma unroll
    for (int v = 0; v < 8; ++v) hp[(long long)v * OUT_DIM] = acc[v];
}

// s_src[i] = h[i,:]·a_s ; s_dst[i] = h[i,:]·a_d   (h row read as 16x float4)
__global__ void k_scores(const float* __restrict__ h, const float* __restrict__ as,
                         const float* __restrict__ ad, float* __restrict__ ss,
                         float* __restrict__ sd, int n) {
    int i = blockIdx.x * blockDim.x + threadIdx.x;
    if (i >= n) return;
    const float4* hr = (const float4*)(h + (long long)i * OUT_DIM);
    const float4* a4 = (const float4*)as;
    const float4* d4 = (const float4*)ad;
    float a = 0.f, b = 0.f;
#pragma unroll
    for (int k = 0; k < OUT_DIM / 4; ++k) {
        float4 v = hr[k], wa = a4[k], wd = d4[k];
        a += v.x * wa.x + v.y * wa.y + v.z * wa.z + v.w * wa.w;
        b += v.x * wd.x + v.y * wd.y + v.z * wd.z + v.w * wd.w;
    }
    ss[i] = a; sd[i] = b;
}

__global__ void k_fill(float* __restrict__ p, long long cnt, float v) {
    long long i = blockIdx.x * (long long)blockDim.x + threadIdx.x;
    if (i < cnt) p[i] = v;
}

// m = -inf, z = 0 in one launch
__global__ void k_init_mz(float* __restrict__ m, float* __restrict__ z, int n) {
    int i = blockIdx.x * blockDim.x + threadIdx.x;
    if (i >= n) return;
    m[i] = -__builtin_inff();
    z[i] = 0.f;
}

__device__ __forceinline__ void edge_sd(const int* __restrict__ ei, long long E,
                                        long long idx, int& s, int& d) {
    if (idx < E) { s = ei[idx]; d = ei[E + idx]; }
    else         { s = d = (int)(idx - E); }       // virtual self-loops
}

__device__ __forceinline__ float lrelu(float v) {
    return v > 0.f ? v : NEG_SLOPE * v;
}

// pass 1: m[dst] = max over incoming edges of leaky_relu(score)
__global__ void k_edge_max(const int* __restrict__ ei, long long E, int n,
                           const float* __restrict__ ss, const float* __restrict__ sd,
                           float* __restrict__ m) {
    long long i = blockIdx.x * (long long)blockDim.x + threadIdx.x;
    if (i >= E + n) return;
    int s, d; edge_sd(ei, E, i, s, d);
    atomicMax(&m[d], lrelu(ss[s] + sd[d]));        // global_atomic_max_num_f32
}

// pass 2: z[dst] += exp(e - m[dst])
__global__ void k_edge_sum(const int* __restrict__ ei, long long E, int n,
                           const float* __restrict__ ss, const float* __restrict__ sd,
                           const float* __restrict__ m, float* __restrict__ z) {
    long long i = blockIdx.x * (long long)blockDim.x + threadIdx.x;
    if (i >= E + n) return;
    int s, d; edge_sd(ei, E, i, s, d);
    atomicAdd(&z[d], __expf(lrelu(ss[s] + sd[d]) - m[d]));
}

// pass 3: materialize alpha per edge once (avoids 8x recompute in aggregate)
__global__ void k_edge_alpha(const int* __restrict__ ei, long long E, int n,
                             const float* __restrict__ ss, const float* __restrict__ sd,
                             const float* __restrict__ m, const float* __restrict__ z,
                             float* __restrict__ alpha) {
    long long i = blockIdx.x * (long long)blockDim.x + threadIdx.x;
    if (i >= E + n) return;
    int s, d; edge_sd(ei, E, i, s, d);
    alpha[i] = __expf(lrelu(ss[s] + sd[d]) - m[d]) / z[d];
}

// pass 4: xnext[dst, colOff:colOff+64] += alpha[e] * h[src,:]
// one thread = one edge x 8 channels: 2x float4 load + 8x global_atomic_add_f32
__global__ void k_edge_agg(const int* __restrict__ ei, long long E, int n,
                           const float* __restrict__ alpha,
                           const float* __restrict__ h, float* __restrict__ xnext,
                           int colOff) {
    long long idx = blockIdx.x * (long long)blockDim.x + threadIdx.x;
    long long tot = (E + n) * 8LL;
    if (idx >= tot) return;
    long long e = idx >> 3;
    int c = (int)(idx & 7) * 8;
    int s, d; edge_sd(ei, E, e, s, d);
    float a = alpha[e];
    const float4* hs = (const float4*)(h + (long long)s * OUT_DIM + c);
    float4 h0 = hs[0], h1 = hs[1];
    float* o = xnext + (long long)d * (2 * OUT_DIM) + colOff + c;
    atomicAdd(&o[0], a * h0.x); atomicAdd(&o[1], a * h0.y);
    atomicAdd(&o[2], a * h0.z); atomicAdd(&o[3], a * h0.w);
    atomicAdd(&o[4], a * h1.x); atomicAdd(&o[5], a * h1.y);
    atomicAdd(&o[6], a * h1.z); atomicAdd(&o[7], a * h1.w);
}

// x = relu(x + bias[col % 64]) in place over [n, 128]
__global__ void k_bias_relu(float* __restrict__ xn, const float* __restrict__ bias,
                            long long cnt) {
    long long i = blockIdx.x * (long long)blockDim.x + threadIdx.x;
    if (i >= cnt) return;
    float v = xn[i] + bias[(int)(i & 127) & 63];
    xn[i] = v > 0.f ? v : 0.f;
}

// g[batch[node], :] += x[node, :]
__global__ void k_pool(const float* __restrict__ x, const int* __restrict__ batch,
                       float* __restrict__ g, long long cnt) {
    long long i = blockIdx.x * (long long)blockDim.x + threadIdx.x;
    if (i >= cnt) return;
    int node = (int)(i >> 7), j = (int)(i & 127);
    atomicAdd(&g[(long long)batch[node] * 128 + j], x[i]);
}

// y[graph] = g[graph,:] · final_w + final_b
__global__ __launch_bounds__(128) void k_final(const float* __restrict__ g,
                                               const float* __restrict__ fw,
                                               const float* __restrict__ fb,
                                               float* __restrict__ y) {
    __shared__ float red[128];
    int t = threadIdx.x;
    red[t] = g[(long long)blockIdx.x * 128 + t] * fw[t];
    __syncthreads();
    for (int sft = 64; sft > 0; sft >>= 1) {
        if (t < sft) red[t] += red[t + sft];
        __syncthreads();
    }
    if (t == 0) y[blockIdx.x] = red[0] + fb[0];
}

// ---------------------------------------------------------------------------
extern "C" void kernel_launch(void* const* d_in, const int* in_sizes, int n_in,
                              void* d_out, int out_size, void* d_ws, size_t ws_size,
                              hipStream_t stream) {
    const float* x_in    = (const float*)d_in[0];
    const int*   ei_sp   = (const int*)d_in[1];
    const int*   ei_dn   = (const int*)d_in[2];
    const int*   batch   = (const int*)d_in[3];
    const float* lin_w   = (const float*)d_in[4];   // [3,128,64]
    const float* att_src = (const float*)d_in[5];   // [3,64]
    const float* att_dst = (const float*)d_in[6];   // [3,64]
    const float* bias    = (const float*)d_in[7];   // [3,64]
    const float* final_w = (const float*)d_in[8];   // [128,1]
    const float* final_b = (const float*)d_in[9];   // [1]
    float*       y       = (float*)d_out;

    const int       n    = in_sizes[0] / IN_DIM;
    const int       nPad = (n + TILE_M - 1) & ~(TILE_M - 1);   // padded rows for h
    const long long Esp  = in_sizes[1] / 2;
    const long long Edn  = in_sizes[2] / 2;
    const long long Emx  = (Esp > Edn ? Esp : Edn) + n;
    const int       ngraphs = out_size;

    // workspace carve-out (~78 MB)
    char* wsb = (char*)d_ws;
    size_t off = 0;
    auto carve = [&](size_t bytes) {
        void* p = wsb + off;
        off = (off + bytes + 255) & ~(size_t)255;
        return p;
    };
    float* h   = (float*)carve((size_t)nPad * OUT_DIM * 4);   // padded: unguarded stores
    float* ss  = (float*)carve((size_t)n * 4);
    float* sd  = (float*)carve((size_t)n * 4);
    float* m   = (float*)carve((size_t)n * 4);
    float* z   = (float*)carve((size_t)n * 4);
    float* al  = (float*)carve((size_t)Emx * 4);
    float* xb0 = (float*)carve((size_t)n * IN_DIM * 4);
    float* xb1 = (float*)carve((size_t)n * IN_DIM * 4);
    float* g   = (float*)carve((size_t)ngraphs * 128 * 4);
    (void)ws_size; (void)n_in;

    const int B = 256;
    auto blks = [](long long work, int b) { return (unsigned)((work + b - 1) / b); };

    for (int l = 0; l < 3; ++l) {
        const float* xcur  = (l == 0) ? x_in : (l == 1 ? xb0 : xb1);
        float*       xnext = (l == 1) ? xb1 : xb0;

        // shared feature transform + attention scores (both branches share W)
        k_gemm_h<<<blks(nPad, TILE_M), 256, 0, stream>>>(xcur, lin_w + (size_t)l * IN_DIM * OUT_DIM, h, n);
        k_scores<<<blks(n, B), B, 0, stream>>>(h, att_src + l * OUT_DIM, att_dst + l * OUT_DIM, ss, sd, n);
        k_fill<<<blks((long long)n * IN_DIM, B), B, 0, stream>>>(xnext, (long long)n * IN_DIM, 0.f);

        const int*      eis[2]  = { ei_sp, ei_dn };
        const long long Es[2]   = { Esp, Edn };
        const int       coff[2] = { 0, OUT_DIM };
        for (int b = 0; b < 2; ++b) {
            long long tot = Es[b] + n;
            k_init_mz<<<blks(n, B), B, 0, stream>>>(m, z, n);
            k_edge_max<<<blks(tot, B), B, 0, stream>>>(eis[b], Es[b], n, ss, sd, m);
            k_edge_sum<<<blks(tot, B), B, 0, stream>>>(eis[b], Es[b], n, ss, sd, m, z);
            k_edge_alpha<<<blks(tot, B), B, 0, stream>>>(eis[b], Es[b], n, ss, sd, m, z, al);
            k_edge_agg<<<blks(tot * 8, B), B, 0, stream>>>(eis[b], Es[b], n, al, h, xnext, coff[b]);
        }
        k_bias_relu<<<blks((long long)n * IN_DIM, B), B, 0, stream>>>(xnext, bias + l * OUT_DIM, (long long)n * IN_DIM);
    }

    // global add pool + final linear (layer 2 wrote xb0)
    k_fill<<<blks((long long)ngraphs * 128, B), B, 0, stream>>>(g, (long long)ngraphs * 128, 0.f);
    k_pool<<<blks((long long)n * 128, B), B, 0, stream>>>(xb0, batch, g, (long long)n * 128);
    k_final<<<ngraphs, 128, 0, stream>>>(g, final_w, final_b, y);
}
